// GAT_28157805592772
// MI455X (gfx1250) — matrix operand order repaired and checked by reading, not compiled, not run on hardware
//
#include <hip/hip_runtime.h>
#include <hip/hip_bf16.h>
#include <stdint.h>

// ---------------------------------------------------------------------------
// GAT 2-layer forward for gfx1250 (MI455X).  GEMMs use v_wmma_f32_16x16x32_f16
// (wave32 WMMA), edge softmax/aggregation use L2-resident float atomics.
// ---------------------------------------------------------------------------

typedef __attribute__((ext_vector_type(16))) _Float16 v16h;
typedef __attribute__((ext_vector_type(8)))  _Float16 v8h;
typedef __attribute__((ext_vector_type(8)))  float    v8f;

#define LRELU_SLOPE 0.2f

// ---------------------------------------------------------------------------
// C[M,N] = A[M,K] * B[N,K]^T   (A,B fp16 row-major, C fp32 row-major)
// One wave -> 16 rows x 64 cols.  Block = 128 threads = 4 waves = 64 rows.
// Requires K % 32 == 0, N % 64 == 0 per grid.y tile, 32B-aligned pointers.
// ---------------------------------------------------------------------------
__global__ __launch_bounds__(128)
void gemm_wmma_f16(const _Float16* __restrict__ A,
                   const _Float16* __restrict__ B,
                   float* __restrict__ C,
                   int M, int N, int K) {
  const int lane    = threadIdx.x & 31;
  const int wave    = threadIdx.x >> 5;
  const int rowTile = blockIdx.x * 4 + wave;        // 16-row tile index
  if (rowTile * 16 >= M) return;                    // wave-uniform exit
  const int colBase = blockIdx.y * 64;

  const int r    = lane & 15;
  const int kSel = lane >> 4;                       // 0 or 1 (half-wave)

  // A-matrix 16x32 fp16 layout: lane m(<16): K = {0..7,16..23}; lane m+16: {8..15,24..31}
  const _Float16* Arow = A + (size_t)(rowTile * 16 + r) * K + kSel * 8;

  v8f acc[4];
#pragma unroll
  for (int t = 0; t < 4; ++t)
#pragma unroll
    for (int v = 0; v < 8; ++v) acc[t][v] = 0.0f;

  for (int k0 = 0; k0 < K; k0 += 32) {
    v8h alo = *(const v8h*)(Arow + k0);        // K = kSel*8 + [0..7]
    v8h ahi = *(const v8h*)(Arow + k0 + 16);   // K = kSel*8 + [16..23]
    v16h a;
#pragma unroll
    for (int j = 0; j < 8; ++j) { a[j] = alo[j]; a[8 + j] = ahi[j]; }

#pragma unroll
    for (int t = 0; t < 4; ++t) {
      // B-matrix 32x16: lane n(<16) holds column n, K = {0..7,16..23}; lane n+16: {8..15,24..31}
      const _Float16* Bp =
          B + (size_t)(colBase + t * 16 + r) * K + k0 + kSel * 8;
      v8h blo = *(const v8h*)Bp;
      v8h bhi = *(const v8h*)(Bp + 16);
      v16h b;
#pragma unroll
      for (int j = 0; j < 8; ++j) { b[j] = blo[j]; b[8 + j] = bhi[j]; }
      acc[t] = __builtin_amdgcn_wmma_f32_16x16x32_f16(
          /*neg_a=*/false, a, /*neg_b=*/false, b,
          /*c_mod=*/(short)0, acc[t], /*reuse_a=*/false, /*reuse_b=*/false);
    }
  }

  // D layout: lane<16 -> VGPR v is row v; lane>=16 -> row v+8; col = lane&15
  const int orow = rowTile * 16 + kSel * 8;
#pragma unroll
  for (int t = 0; t < 4; ++t)
#pragma unroll
    for (int v = 0; v < 8; ++v)
      C[(size_t)(orow + v) * N + colBase + t * 16 + r] = acc[t][v];
}

// ---------------------------------------------------------------------------
// Elementwise helpers
// ---------------------------------------------------------------------------
__global__ void cvt_f32_to_f16(const float* __restrict__ in,
                               _Float16* __restrict__ out, int n) {
  for (int i = blockIdx.x * blockDim.x + threadIdx.x; i < n;
       i += gridDim.x * blockDim.x)
    out[i] = (_Float16)in[i];
}

__global__ void fill_f32(float* p, float v, int n) {
  for (int i = blockIdx.x * blockDim.x + threadIdx.x; i < n;
       i += gridDim.x * blockDim.x)
    p[i] = v;
}

__global__ void fill_u32(unsigned* p, unsigned v, int n) {
  for (int i = blockIdx.x * blockDim.x + threadIdx.x; i < n;
       i += gridDim.x * blockDim.x)
    p[i] = v;
}

// ---------------------------------------------------------------------------
// Per-node attention coefficients: as[n]=h[n,:].a_src, ad[n]=h[n,:].a_dst
// One wave per node.
// ---------------------------------------------------------------------------
__global__ void node_alphas(const float* __restrict__ h,
                            const float* __restrict__ a_s,
                            const float* __restrict__ a_d,
                            float* __restrict__ as, float* __restrict__ ad,
                            int n_nodes, int F) {
  int node = (blockIdx.x * blockDim.x + threadIdx.x) >> 5;
  int lane = threadIdx.x & 31;
  if (node >= n_nodes) return;
  float s = 0.f, d = 0.f;
  for (int f = lane; f < F; f += 32) {
    float v = h[(size_t)node * F + f];
    s += v * a_s[f];
    d += v * a_d[f];
  }
  for (int off = 16; off; off >>= 1) {
    s += __shfl_down(s, off, 32);
    d += __shfl_down(d, off, 32);
  }
  if (lane == 0) { as[node] = s; ad[node] = d; }
}

// Monotone uint key for float atomicMax (handles negatives)
__device__ __forceinline__ unsigned f32_to_key(float f) {
  unsigned u = __float_as_uint(f);
  return (u & 0x80000000u) ? ~u : (u | 0x80000000u);
}
__device__ __forceinline__ float key_to_f32(unsigned k) {
  unsigned u = (k & 0x80000000u) ? (k & 0x7FFFFFFFu) : ~k;
  return __uint_as_float(u);
}

// Pass 1: e = leaky_relu(as[src]+ad[dst]); store e; atomicMax key into m[dst]
__global__ void edge_max(const int* __restrict__ es, const int* __restrict__ ed,
                         const float* __restrict__ as,
                         const float* __restrict__ ad,
                         float* __restrict__ ebuf, unsigned* __restrict__ mkey,
                         int E, int n_nodes) {
  int total = E + n_nodes;   // implicit self-loops appended
  for (int i = blockIdx.x * blockDim.x + threadIdx.x; i < total;
       i += gridDim.x * blockDim.x) {
    int s = (i < E) ? es[i] : (i - E);
    int d = (i < E) ? ed[i] : (i - E);
    float v = as[s] + ad[d];
    float e = (v > 0.f) ? v : LRELU_SLOPE * v;
    ebuf[i] = e;
    atomicMax(&mkey[d], f32_to_key(e));
  }
}

__global__ void decode_keys(float* m, int n) {   // in-place key -> float
  unsigned* mk = (unsigned*)m;
  for (int i = blockIdx.x * blockDim.x + threadIdx.x; i < n;
       i += gridDim.x * blockDim.x)
    m[i] = key_to_f32(mk[i]);
}

// Pass 2: ee = exp(e - m[dst]); overwrite ebuf; denom[dst] += ee
__global__ void edge_expsum(const int* __restrict__ ed,
                            const float* __restrict__ m,
                            float* __restrict__ ebuf,
                            float* __restrict__ denom, int E, int n_nodes) {
  int total = E + n_nodes;
  for (int i = blockIdx.x * blockDim.x + threadIdx.x; i < total;
       i += gridDim.x * blockDim.x) {
    int d = (i < E) ? ed[i] : (i - E);
    float ee = __expf(ebuf[i] - m[d]);
    ebuf[i] = ee;
    atomicAdd(&denom[d], ee);
  }
}

// Pass 3: wave per edge; agg[dst,:] += (ee/denom[dst]) * h[src,:]
__global__ void edge_aggregate(const int* __restrict__ es,
                               const int* __restrict__ ed,
                               const float* __restrict__ ebuf,
                               const float* __restrict__ denom,
                               const float* __restrict__ h,
                               float* __restrict__ agg, int E, int n_nodes,
                               int F) {
  int lane  = threadIdx.x & 31;
  int waves = (gridDim.x * blockDim.x) >> 5;
  int total = E + n_nodes;
  for (int i = (blockIdx.x * blockDim.x + threadIdx.x) >> 5; i < total;
       i += waves) {
    int s = (i < E) ? es[i] : (i - E);
    int d = (i < E) ? ed[i] : (i - E);
    float alpha = ebuf[i] / denom[d];
    const float* hs = h + (size_t)s * F;
    float* od = agg + (size_t)d * F;
    for (int f = lane; f < F; f += 32)
      atomicAdd(&od[f], alpha * hs[f]);
  }
}

// Layer-1 epilogue: h1a = (fp16) relu(agg + b)
__global__ void epilogue_relu_f16(const float* __restrict__ agg,
                                  const float* __restrict__ b,
                                  _Float16* __restrict__ out, int n, int F) {
  for (int i = blockIdx.x * blockDim.x + threadIdx.x; i < n;
       i += gridDim.x * blockDim.x) {
    float v = agg[i] + b[i % F];
    out[i] = (_Float16)(v > 0.f ? v : 0.f);
  }
}

// Layer-2 epilogue: out = agg + b (fp32, final result)
__global__ void epilogue_bias(const float* __restrict__ agg,
                              const float* __restrict__ b,
                              float* __restrict__ out, int n, int F) {
  for (int i = blockIdx.x * blockDim.x + threadIdx.x; i < n;
       i += gridDim.x * blockDim.x)
    out[i] = agg[i] + b[i % F];
}

// ---------------------------------------------------------------------------
// Host side
// ---------------------------------------------------------------------------
static inline size_t align256(size_t x) { return (x + 255) & ~(size_t)255; }

extern "C" void kernel_launch(void* const* d_in, const int* in_sizes, int n_in,
                              void* d_out, int out_size, void* d_ws,
                              size_t ws_size, hipStream_t stream) {
  // Inputs (setup_inputs order)
  const float* x     = (const float*)d_in[0];
  const int*   eidx  = (const int*)d_in[1];
  const float* W1    = (const float*)d_in[2];
  const float* asrc1 = (const float*)d_in[3];
  const float* adst1 = (const float*)d_in[4];
  const float* b1    = (const float*)d_in[5];
  const float* W2    = (const float*)d_in[6];
  const float* asrc2 = (const float*)d_in[7];
  const float* adst2 = (const float*)d_in[8];
  const float* b2    = (const float*)d_in[9];
  float* out = (float*)d_out;

  const int HID   = in_sizes[3];             // 128
  const int OUTF  = in_sizes[7];             // 64
  const int INDIM = in_sizes[2] / HID;       // 256
  const int N     = in_sizes[0] / INDIM;     // 50000
  const int E     = in_sizes[1] / 2;         // 800000
  const int ET    = E + N;                   // with self-loops
  const int* es = eidx;                      // edge_index[0]
  const int* ed = eidx + E;                  // edge_index[1]

  // Workspace layout (regions reused across phases)
  char* ws = (char*)d_ws;
  size_t off = 0;
  size_t szR0 = align256((size_t)N * INDIM * sizeof(_Float16)); // == N*HID*4
  size_t szR1 = align256((size_t)N * HID * sizeof(float));
  size_t szR2 = align256((size_t)N * HID * sizeof(_Float16));
  size_t szEB = align256((size_t)ET * sizeof(float));
  size_t szW1 = align256((size_t)HID * INDIM * sizeof(_Float16));
  size_t szW2 = align256((size_t)OUTF * HID * sizeof(_Float16));
  size_t szNV = align256((size_t)N * sizeof(float));

  char* R0   = ws + off; off += szR0;  // x_f16  -> later agg1 (f32)
  char* R1   = ws + off; off += szR1;  // h1 f32 -> later h2 (f32) + agg2 (f32)
  char* R2   = ws + off; off += szR2;  // h1 activated, f16 (layer-2 GEMM input)
  char* EB   = ws + off; off += szEB;  // per-edge e / ee
  char* W1H  = ws + off; off += szW1;
  char* W2H  = ws + off; off += szW2;
  char* AS   = ws + off; off += szNV;
  char* AD   = ws + off; off += szNV;
  char* MB   = ws + off; off += szNV;  // max keys -> floats
  char* DEN  = ws + off; off += szNV;
  (void)ws_size; (void)n_in; (void)out_size;

  _Float16* xh   = (_Float16*)R0;
  float*    agg1 = (float*)R0;
  float*    h1   = (float*)R1;
  float*    h2   = (float*)R1;
  float*    agg2 = (float*)(R1 + align256((size_t)N * OUTF * sizeof(float)));
  _Float16* h1a  = (_Float16*)R2;
  float*    ebuf = (float*)EB;
  _Float16* w1h  = (_Float16*)W1H;
  _Float16* w2h  = (_Float16*)W2H;
  float *as = (float*)AS, *ad = (float*)AD, *m = (float*)MB, *den = (float*)DEN;

  const int TB = 256;
  const int rowBlocks = (N / 16 + 3) / 4 + 1;   // 16-row tiles, 4 waves/block
  const int edgeBlocks = (ET + TB - 1) / TB;
  const int waveBlocks = ((size_t)ET * 32 + TB - 1) / TB;
  const int nodeWaveBlocks = ((size_t)N * 32 + TB - 1) / TB;

  // ---- conversions ----
  cvt_f32_to_f16<<<2048, TB, 0, stream>>>(x, xh, N * INDIM);
  cvt_f32_to_f16<<<64, TB, 0, stream>>>(W1, w1h, HID * INDIM);
  cvt_f32_to_f16<<<32, TB, 0, stream>>>(W2, w2h, OUTF * HID);

  // ================= Layer 1 =================
  gemm_wmma_f16<<<dim3(rowBlocks, HID / 64), 128, 0, stream>>>(
      xh, w1h, h1, N, HID, INDIM);
  node_alphas<<<nodeWaveBlocks, TB, 0, stream>>>(h1, asrc1, adst1, as, ad, N, HID);

  fill_f32<<<2048, TB, 0, stream>>>(agg1, 0.f, N * HID);
  fill_f32<<<256, TB, 0, stream>>>(den, 0.f, N);
  fill_u32<<<256, TB, 0, stream>>>((unsigned*)m, 0u, N);

  edge_max<<<edgeBlocks, TB, 0, stream>>>(es, ed, as, ad, ebuf, (unsigned*)m, E, N);
  decode_keys<<<256, TB, 0, stream>>>(m, N);
  edge_expsum<<<edgeBlocks, TB, 0, stream>>>(ed, m, ebuf, den, E, N);
  edge_aggregate<<<waveBlocks, TB, 0, stream>>>(es, ed, ebuf, den, h1, agg1, E, N, HID);
  epilogue_relu_f16<<<2048, TB, 0, stream>>>(agg1, b1, h1a, N * HID, HID);

  // ================= Layer 2 =================
  gemm_wmma_f16<<<dim3(rowBlocks, OUTF / 64), 128, 0, stream>>>(
      h1a, w2h, h2, N, OUTF, HID);
  node_alphas<<<nodeWaveBlocks, TB, 0, stream>>>(h2, asrc2, adst2, as, ad, N, OUTF);

  fill_f32<<<2048, TB, 0, stream>>>(agg2, 0.f, N * OUTF);
  fill_f32<<<256, TB, 0, stream>>>(den, 0.f, N);
  fill_u32<<<256, TB, 0, stream>>>((unsigned*)m, 0u, N);

  edge_max<<<edgeBlocks, TB, 0, stream>>>(es, ed, as, ad, ebuf, (unsigned*)m, E, N);
  decode_keys<<<256, TB, 0, stream>>>(m, N);
  edge_expsum<<<edgeBlocks, TB, 0, stream>>>(ed, m, ebuf, den, E, N);
  edge_aggregate<<<waveBlocks, TB, 0, stream>>>(es, ed, ebuf, den, h2, agg2, E, N, OUTF);
  epilogue_bias<<<2048, TB, 0, stream>>>(agg2, b2, out, N * OUTF, OUTF);
}
